// NonIsotropicVMF_26680336843243
// MI455X (gfx1250) — matrix operand order repaired and checked by reading, not compiled, not run on hardware
//
#include <hip/hip_runtime.h>
#include <hip/hip_bf16.h>
#include <math.h>

// ---------------------------------------------------------------------------
// Problem constants (match reference)
// ---------------------------------------------------------------------------
#define BB   512    // batch
#define CC   8192   // classes
#define DD   512    // feature dim
#define NS   16     // n_samples
#define BUDG 8      // rejection budget
#define NK   (DD / 32)   // 16 K-steps of 32

typedef __attribute__((ext_vector_type(16))) __bf16 v16bf;
typedef __attribute__((ext_vector_type(8)))  float  v8f;
typedef __attribute__((ext_vector_type(4)))  int    v4i;

// ---------------------------------------------------------------------------
// gfx1250 async global->LDS path (guarded; falls back to sync stage)
// ---------------------------------------------------------------------------
#if defined(__has_builtin)
#  if __has_builtin(__builtin_amdgcn_global_load_async_to_lds_b128) && \
      __has_builtin(__builtin_amdgcn_s_wait_asynccnt)
#    define USE_ASYNC_LDS 1
#  endif
#endif
#ifndef USE_ASYNC_LDS
#  define USE_ASYNC_LDS 0
#endif

#if USE_ASYNC_LDS
__device__ __forceinline__ void async_cp16(const void* g, void* l) {
    __builtin_amdgcn_global_load_async_to_lds_b128(
        (__attribute__((address_space(1))) v4i*)(g),
        (__attribute__((address_space(3))) v4i*)(l),
        0, 0);
}
#endif

// ---------------------------------------------------------------------------
// Helpers
// ---------------------------------------------------------------------------
__device__ __forceinline__ unsigned short f2bf(float f) {
    unsigned u = __float_as_uint(f);
    unsigned r = u + 0x7FFFu + ((u >> 16) & 1u);   // round-to-nearest-even
    return (unsigned short)(r >> 16);
}

__device__ __forceinline__ unsigned pcg(unsigned v) {
    v = v * 747796405u + 2891336453u;
    unsigned x = ((v >> ((v >> 28u) + 4u)) ^ v) * 277803737u;
    return (x >> 22u) ^ x;
}
__device__ __forceinline__ unsigned hash3(unsigned a, unsigned b, unsigned c) {
    return pcg(a + pcg(b + pcg(c)));
}
__device__ __forceinline__ float u01(unsigned h) {
    return ((float)(h >> 8) + 0.5f) * (1.0f / 16777216.0f);   // (0,1)
}

__device__ __forceinline__ float blockReduceSum(float v, float* red, int tid) {
    #pragma unroll
    for (int o = 16; o > 0; o >>= 1) v += __shfl_xor(v, o, 32);
    if ((tid & 31) == 0) red[tid >> 5] = v;
    __syncthreads();
    if (tid < 8) {
        float x = red[tid];
        #pragma unroll
        for (int o = 4; o > 0; o >>= 1) x += __shfl_xor(x, o, 32);
        if (tid == 0) red[0] = x;
    }
    __syncthreads();
    float r = red[0];
    __syncthreads();
    return r;
}

__device__ __forceinline__ float blockReduceMax(float v, float* red, int tid) {
    #pragma unroll
    for (int o = 16; o > 0; o >>= 1) v = fmaxf(v, __shfl_xor(v, o, 32));
    if ((tid & 31) == 0) red[tid >> 5] = v;
    __syncthreads();
    if (tid < 8) {
        float x = red[tid];
        #pragma unroll
        for (int o = 4; o > 0; o >>= 1) x = fmaxf(x, __shfl_xor(x, o, 32));
        if (tid == 0) red[0] = x;
    }
    __syncthreads();
    float r = red[0];
    __syncthreads();
    return r;
}

// ---------------------------------------------------------------------------
// K0: zero output scalar
// ---------------------------------------------------------------------------
__global__ void vmf_zero(float* o) { o[0] = 0.0f; }

// ---------------------------------------------------------------------------
// K1: per-class prep.  One block per class c (256 threads, 2 elems/thread).
// ---------------------------------------------------------------------------
__global__ void vmf_classprep(const float* __restrict__ cw,
                              const float* __restrict__ kw,
                              unsigned short* __restrict__ mu2n,
                              float* __restrict__ norm_mu2,
                              float* __restrict__ bias) {
    __shared__ float red[8];
    const int c = blockIdx.x;
    const int t = threadIdx.x;
    const float c0 = cw[c * DD + t];
    const float c1 = cw[c * DD + t + 256];
    const float inv_nc = rsqrtf(blockReduceSum(c0 * c0 + c1 * c1, red, t));
    const float k0 = fmaxf(kw[c * DD + t],       0.1f);
    const float k1 = fmaxf(kw[c * DD + t + 256], 0.1f);
    const float m0 = c0 * inv_nc * k0;
    const float m1 = c1 * inv_nc * k1;
    const float n2 = blockReduceSum(m0 * m0 + m1 * m1, red, t);
    const float nm = sqrtf(n2);
    const float slk = blockReduceSum(__logf(k0) + __logf(k1), red, t);
    const float inm = 1.0f / nm;
    mu2n[c * DD + t]       = f2bf(m0 * inm);
    mu2n[c * DD + t + 256] = f2bf(m1 * inm);
    if (t == 0) {
        const float nu = (float)(DD / 2 - 1);              // 255
        const float L2PI = 1.8378770664093453f;            // log(2*pi)
        const float s = sqrtf(nu * nu + nm * nm);
        const float logiv = s + nu * __logf(nm) - nu * __logf(nu + s)
                            - 0.5f * L2PI - 0.25f * __logf(nu * nu + nm * nm);
        const float logcp = nu * __logf(nm) - (float)(DD / 2) * L2PI - logiv;
        norm_mu2[c] = nm;
        bias[c] = logcp + slk - 2.0f * __logf(nm);
    }
}

// ---------------------------------------------------------------------------
// K2: vMF sampling (Wood's alg, fixed budget, deterministic hash PRNG).
// ---------------------------------------------------------------------------
__global__ void vmf_sample(const float* __restrict__ feat,
                           const float* __restrict__ unc,
                           unsigned short* __restrict__ samples) {
    __shared__ float red[8];
    const int nb = blockIdx.x;
    const int n = nb & (NS - 1);
    const int b = nb >> 4;
    const int t = threadIdx.x;

    const float f0 = feat[b * DD + t];
    const float f1 = feat[b * DD + t + 256];
    const float inf_ = rsqrtf(blockReduceSum(f0 * f0 + f1 * f1, red, t));
    const float mu0 = f0 * inf_, mu1 = f1 * inf_;
    const float kappa = 1.0f / unc[b];

    const float dm1 = (float)(DD - 1);                   // 511
    const float sq = sqrtf(4.0f * kappa * kappa + dm1 * dm1);
    const float bw = (-2.0f * kappa + sq) / dm1;
    const float aw = (dm1 + 2.0f * kappa + sq) * 0.25f;
    const float d0c = 4.0f * aw * bw / (1.0f + bw) - dm1 * __logf(dm1);

    float w = 0.0f; bool found = false;
    #pragma unroll
    for (int i = 0; i < BUDG; ++i) {
        const float u1 = u01(hash3(nb, (unsigned)i, 0x9E3779B9u));
        const float u2 = u01(hash3(nb, (unsigned)i, 0x85EBCA6Bu));
        const float u3 = u01(hash3(nb, (unsigned)i, 0xC2B2AE35u));
        float g = sqrtf(-2.0f * __logf(u1)) * __cosf(6.2831853f * u2);
        float z = 0.5f + g * 0.02209709f;                // Beta(a,a) normal approx
        z = fminf(fmaxf(z, 1e-6f), 1.0f - 1e-6f);
        const float den = 1.0f - (1.0f - bw) * z;
        const float wi = (1.0f - (1.0f + bw) * z) / den;
        const float tt = 2.0f * aw * bw / den;
        const bool acc = (dm1 * __logf(tt) - tt + d0c) >= __logf(u3);
        if (i == 0) w = wi;
        if (!found && acc) { w = wi; found = true; }
    }

    const float ua0 = u01(hash3(nb, (unsigned)t,         0x27220A95u));
    const float ub0 = u01(hash3(nb, (unsigned)t,         0x165667B1u));
    const float ua1 = u01(hash3(nb, (unsigned)(t + 256), 0x27220A95u));
    const float ub1 = u01(hash3(nb, (unsigned)(t + 256), 0x165667B1u));
    float g0 = sqrtf(-2.0f * __logf(ua0)) * __cosf(6.2831853f * ub0);
    float g1 = sqrtf(-2.0f * __logf(ua1)) * __cosf(6.2831853f * ub1);
    const float dot = blockReduceSum(g0 * mu0 + g1 * mu1, red, t);
    float v0 = g0 - dot * mu0;
    float v1 = g1 - dot * mu1;
    const float inv_nv = rsqrtf(blockReduceSum(v0 * v0 + v1 * v1, red, t) + 1e-20f);
    const float sw = sqrtf(fmaxf(0.0f, 1.0f - w * w));
    unsigned short* row = samples + (size_t)(b * NS + n) * DD;
    row[t]       = f2bf(w * mu0 + sw * v0 * inv_nv);
    row[t + 256] = f2bf(w * mu1 + sw * v1 * inv_nv);
}

// ---------------------------------------------------------------------------
// K3: fused bf16 WMMA GEMM + logsumexp over the 16 samples.
//   Tile 128 rows (8 b) x 128 cols, 256 thr = 8 waves (4 M-groups x 2 N-halves).
//   Wave: 2 b-groups x 64 cols -> 8 x V_WMMA_F32_16X16X32_BF16 per K-step.
//   Double-buffered LDS (32 KB); async global->LDS DMA when available.
// ---------------------------------------------------------------------------
__global__ void __launch_bounds__(256)
vmf_gemm_lse(const unsigned short* __restrict__ S,
             const unsigned short* __restrict__ W,
             const float* __restrict__ norm_mu2,
             const float* __restrict__ bias,
             float* __restrict__ sim) {
    // buf layout (shorts): A0[0,4096) B0[4096,8192) A1[8192,12288) B1[12288,16384)
    __shared__ __align__(16) unsigned short lds[16384];   // 32 KB

    const int tid  = threadIdx.x;
    const int lane = tid & 31;
    const int wave = tid >> 5;
    const int wm   = wave & 3;       // M quarter: rows [wm*32, wm*32+32)
    const int wn   = wave >> 2;      // N half
    const int hi   = lane >> 4;      // lane half
    const int m    = lane & 15;

    const int cBase   = blockIdx.x * 128;
    const int bBase   = blockIdx.y * 8;
    const int rowBase = bBase * NS;  // 128 rows per tile

    v8f acc[2][4];
    #pragma unroll
    for (int g = 0; g < 2; ++g)
        #pragma unroll
        for (int j = 0; j < 4; ++j) {
            v8f z = {0,0,0,0,0,0,0,0};
            acc[g][j] = z;
        }

    // ---- stage issue: 512 x 16B chunks each for A and B; 2 of each / thread
#if USE_ASYNC_LDS
    #define STAGE(buf, k0)                                                        \
        do {                                                                      \
            _Pragma("unroll")                                                     \
            for (int i_ = 0; i_ < 2; ++i_) {                                      \
                const int id_ = tid * 2 + i_;       /* 0..511 = row*4 + q */      \
                const int r_ = id_ >> 2, q_ = id_ & 3;                            \
                async_cp16(S + (size_t)(rowBase + r_) * DD + (k0) + q_ * 8,       \
                           lds + (buf) * 8192 + id_ * 8);                         \
                async_cp16(W + (size_t)(cBase + r_) * DD + (k0) + q_ * 8,         \
                           lds + (buf) * 8192 + 4096 + id_ * 8);                  \
            }                                                                     \
        } while (0)
#else
    #define STAGE(buf, k0)                                                        \
        do {                                                                      \
            _Pragma("unroll")                                                     \
            for (int i_ = 0; i_ < 2; ++i_) {                                      \
                const int id_ = tid * 2 + i_;                                     \
                const int r_ = id_ >> 2, q_ = id_ & 3;                            \
                *(float4*)(lds + (buf) * 8192 + id_ * 8) =                        \
                    *(const float4*)(S + (size_t)(rowBase + r_) * DD + (k0) + q_ * 8); \
                *(float4*)(lds + (buf) * 8192 + 4096 + id_ * 8) =                 \
                    *(const float4*)(W + (size_t)(cBase + r_) * DD + (k0) + q_ * 8);   \
            }                                                                     \
            if ((k0) + 32 < DD) {                                                 \
                __builtin_prefetch(S + (size_t)(rowBase + (tid >> 1)) * DD + (k0) + 32, 0, 1); \
                __builtin_prefetch(W + (size_t)(cBase + (tid >> 1)) * DD + (k0) + 32, 0, 1);   \
            }                                                                     \
        } while (0)
#endif

    STAGE(0, 0);   // prologue: fill buffer 0

    #pragma unroll
    for (int k = 0; k < NK; ++k) {
        const int cur = k & 1;
        if (k + 1 < NK) {
            STAGE(1 - cur, (k + 1) * 32);     // issue next stage into other buf
#if USE_ASYNC_LDS
            __builtin_amdgcn_s_wait_asynccnt(4);   // drain current stage's 4 DMAs
#endif
        } else {
#if USE_ASYNC_LDS
            __builtin_amdgcn_s_wait_asynccnt(0);
#endif
        }
        __syncthreads();                      // stage `cur` visible to all waves

        const unsigned short* Abuf = lds + cur * 8192;
        const unsigned short* Bbuf = Abuf + 4096;

        // A fragments: rows (wm*32 + g*16 + m); lanes 0-15: K{0-7,16-23},
        // lanes 16-31: K{8-15,24-31}
        v16bf afrag[2];
        #pragma unroll
        for (int g = 0; g < 2; ++g) {
            const unsigned short* arow = Abuf + (wm * 32 + g * 16 + m) * 32;
            ((float4*)&afrag[g])[0] = *(const float4*)(arow + hi * 8);
            ((float4*)&afrag[g])[1] = *(const float4*)(arow + 16 + hi * 8);
        }
        // B fragments: one column per lane; lanes 0-15: K{0-7,16-23}, 16-31: K{8-15,24-31}
        #pragma unroll
        for (int j = 0; j < 4; ++j) {
            v16bf bfrag;
            const unsigned short* bcol = Bbuf + (wn * 64 + j * 16 + m) * 32 + hi * 8;
            ((float4*)&bfrag)[0] = *(const float4*)(bcol);
            ((float4*)&bfrag)[1] = *(const float4*)(bcol + 16);
            #pragma unroll
            for (int g = 0; g < 2; ++g)
                acc[g][j] = __builtin_amdgcn_wmma_f32_16x16x32_bf16(
                                false, afrag[g], false, bfrag, (short)0,
                                acc[g][j], false, false);
        }
        __syncthreads();                      // done reading buf before reuse
    }
    #undef STAGE

    // ---- epilogue: logsumexp over M (16 samples of each b) per column ----
    const float LOG_NS = 2.7725887222397811f;   // log(16)
    #pragma unroll
    for (int g = 0; g < 2; ++g) {
        const int b = bBase + wm * 2 + g;
        #pragma unroll
        for (int j = 0; j < 4; ++j) {
            const int c = cBase + wn * 64 + j * 16 + m;
            const float s = norm_mu2[c];
            float v[8], mx = -1e30f;
            #pragma unroll
            for (int kk = 0; kk < 8; ++kk) {
                v[kk] = acc[g][j][kk] * s;
                mx = fmaxf(mx, v[kk]);
            }
            mx = fmaxf(mx, __shfl_xor(mx, 16, 32));      // join lane halves
            float se = 0.0f;
            #pragma unroll
            for (int kk = 0; kk < 8; ++kk) se += __expf(v[kk] - mx);
            se += __shfl_xor(se, 16, 32);
            const float r = bias[c] + mx + __logf(se) - LOG_NS;
            if (lane < 16) sim[(size_t)b * CC + c] = r;
        }
    }
}

// ---------------------------------------------------------------------------
// K4: loss = mean_b( logsumexp_c(sim) - sim[b, y[b]] )
// ---------------------------------------------------------------------------
__global__ void vmf_loss(const float* __restrict__ sim,
                         const int* __restrict__ y,
                         float* __restrict__ out) {
    __shared__ float red[8];
    const int b = blockIdx.x;
    const int t = threadIdx.x;
    float mx = -1e30f;
    for (int c = t; c < CC; c += 256) mx = fmaxf(mx, sim[(size_t)b * CC + c]);
    mx = blockReduceMax(mx, red, t);
    float se = 0.0f;
    for (int c = t; c < CC; c += 256) se += __expf(sim[(size_t)b * CC + c] - mx);
    se = blockReduceSum(se, red, t);
    if (t == 0) {
        const float lse = mx + __logf(se);
        const float pos = sim[(size_t)b * CC + y[b]];
        atomicAdd(out, (lse - pos) * (1.0f / (float)BB));
    }
}

// ---------------------------------------------------------------------------
// Launch
// ---------------------------------------------------------------------------
extern "C" void kernel_launch(void* const* d_in, const int* in_sizes, int n_in,
                              void* d_out, int out_size, void* d_ws, size_t ws_size,
                              hipStream_t stream) {
    const float* unc  = (const float*)d_in[1];
    const int*   y    = (const int*)d_in[2];
    const float* feat = (const float*)d_in[3];
    const float* cw   = (const float*)d_in[4];
    const float* kw   = (const float*)d_in[5];
    float* out = (float*)d_out;

    char* ws = (char*)d_ws;
    unsigned short* mu2n    = (unsigned short*)(ws);                               // 8 MB
    unsigned short* samples = (unsigned short*)(ws + (size_t)8  * 1024 * 1024);    // 8 MB
    float* norm_mu2         = (float*)(ws + (size_t)16 * 1024 * 1024);             // 32 KB
    float* bias             = (float*)(ws + (size_t)16 * 1024 * 1024 + 32 * 1024); // 32 KB
    float* sim              = (float*)(ws + (size_t)16 * 1024 * 1024 + 64 * 1024); // 16 MB

    vmf_zero<<<1, 1, 0, stream>>>(out);
    vmf_classprep<<<CC, 256, 0, stream>>>(cw, kw, mu2n, norm_mu2, bias);
    vmf_sample<<<BB * NS, 256, 0, stream>>>(feat, unc, samples);
    vmf_gemm_lse<<<dim3(CC / 128, BB / 8), 256, 0, stream>>>(samples, mu2n,
                                                             norm_mu2, bias, sim);
    vmf_loss<<<BB, 256, 0, stream>>>(sim, y, out);
}
